// HAN_29437705846841
// MI455X (gfx1250) — compile-verified, matrix operand order checked by reading, not run on hardware
//
#include <hip/hip_runtime.h>
#include <hip/hip_bf16.h>
#include <stdint.h>

// ---------------------------------------------------------------------------
// HAN word encoder for MI455X (gfx1250, wave32, WMMA).
// GEMM precision: bf16 operands, f32 accumulate (v_wmma_f32_16x16x32_bf16).
// GRU hidden state kept fp32 in global memory for recurrence accuracy.
// X tile staged into LDS with global_load_async_to_lds_b128 (ASYNCcnt).
// ---------------------------------------------------------------------------

typedef __attribute__((ext_vector_type(16))) __bf16        bf16x16;
typedef __attribute__((ext_vector_type(8)))  float         f32x8;
typedef __attribute__((ext_vector_type(8)))  unsigned short u16x8;

union FragAB { bf16x16 bf; u16x8 u[2]; };

#define NSEQ 1280      // B*D
#define N2   2560      // fwd + bwd virtual rows
#define TSEQ 50
#define EDIM 300
#define XS   304       // padded embedding width (16B-aligned rows)
#define HDIM 256
#define H2   512
#define KPAD 576       // 304 (x) + 256 (h) + 16 pad  -> 18 k-steps of 32
#define LDA  584       // LDS row stride in halves (1168B = 73*16, bank-safe)
#define MT   32        // rows per workgroup

__device__ __forceinline__ unsigned short f2bf(float f) {
    unsigned int u = __float_as_uint(f);
    unsigned int r = (u + 0x7FFFu + ((u >> 16) & 1u)) >> 16;   // RNE
    return (unsigned short)r;
}
__device__ __forceinline__ float bf2f(unsigned short h) {
    return __uint_as_float(((unsigned int)h) << 16);
}

// Async global->LDS 16B copy (gfx1250).  lds_off is the byte offset relative
// to the workgroup LDS base (dynamic smem starts at offset 0).
__device__ __forceinline__ void async_g2l_b128(unsigned lds_off, const void* gptr) {
    asm volatile("global_load_async_to_lds_b128 %0, %1, off"
                 :: "v"(lds_off), "v"(gptr) : "memory");
}
__device__ __forceinline__ void wait_async0() {
#if __has_builtin(__builtin_amdgcn_s_wait_asynccnt)
    __builtin_amdgcn_s_wait_asynccnt(0);
#else
    asm volatile("s_wait_asynccnt 0x0" ::: "memory");
#endif
}

// Load one 16x32 bf16 operand fragment from a row-major [rows][ld] matrix.
// CDNA5 16-bit A layout: lanes 0-15 hold K={0..7,16..23} of row M=lane,
// lanes 16-31 hold K={8..15,24..31} of row M=lane-16.  B mirrored with N as
// the row (weights stored transposed).
__device__ __forceinline__ bf16x16 load_frag(const unsigned short* base, int ld,
                                             int row0, int k0, int lane) {
    int r  = row0 + (lane & 15);
    int kb = k0 + ((lane >> 4) << 3);
    const unsigned short* p = base + (size_t)r * ld + kb;
    FragAB f;
    f.u[0] = *(const u16x8*)(p);
    f.u[1] = *(const u16x8*)(p + 16);
    return f.bf;
}

__device__ __forceinline__ f32x8 wmma_bf16(bf16x16 a, bf16x16 b, f32x8 c) {
    return __builtin_amdgcn_wmma_f32_16x16x32_bf16(
        false, a, false, b, (short)0, c, false, false);
}

// ---------------------------------------------------------------------------
// Weight packing: Wg[556][512], Wc[556][256], Wp[512][256] (row-major, fp32)
// -> bf16 transposed [n][Kpad] with K map: k<300 -> K=k ; k>=300 -> K=k+4.
// ---------------------------------------------------------------------------
__global__ void k_pack_w(const float* __restrict__ Wg, const float* __restrict__ Wc,
                         const float* __restrict__ Wp,
                         unsigned short* __restrict__ WgT,
                         unsigned short* __restrict__ WcT,
                         unsigned short* __restrict__ WpT) {
    const int NG = H2 * KPAD, NC = HDIM * KPAD, NP = HDIM * H2;
    int idx = blockIdx.x * blockDim.x + threadIdx.x;
    if (idx < NG) {
        int n = idx / KPAD, K = idx % KPAD;
        float v = 0.f;
        if (K < EDIM)                    v = Wg[K * H2 + n];
        else if (K >= XS && K < XS+HDIM) v = Wg[(K - 4) * H2 + n];
        WgT[idx] = f2bf(v);
    } else if (idx < NG + NC) {
        int j = idx - NG;
        int n = j / KPAD, K = j % KPAD;
        float v = 0.f;
        if (K < EDIM)                    v = Wc[K * HDIM + n];
        else if (K >= XS && K < XS+HDIM) v = Wc[(K - 4) * HDIM + n];
        WcT[j] = f2bf(v);
    } else if (idx < NG + NC + NP) {
        int j = idx - NG - NC;
        int n = j / H2, k = j % H2;
        WpT[j] = f2bf(Wp[k * HDIM + n]);
    }
}

// ---------------------------------------------------------------------------
// Embedding gather -> bf16 X[t][n2][XS]; backward rows (n2>=NSEQ) use
// reverse_sequence time indices so one GRU pass handles both directions.
// ---------------------------------------------------------------------------
__global__ void k_gather(const int* __restrict__ ids, const int* __restrict__ lens,
                         const float* __restrict__ Emb, unsigned short* __restrict__ X) {
    int bid = blockIdx.x;                 // t * N2 + n2
    int n2  = bid % N2;
    int t   = bid / N2;
    int seq = (n2 < NSEQ) ? n2 : n2 - NSEQ;
    int tt  = t;
    if (n2 >= NSEQ) {
        int len = lens[seq];
        tt = (t < len) ? (len - 1 - t) : t;
    }
    int token = ids[seq * TSEQ + tt];
    const float* e = Emb + (size_t)token * EDIM;
    unsigned short* xr = X + ((size_t)t * N2 + n2) * XS;
    for (int c = threadIdx.x; c < XS; c += blockDim.x)
        xr[c] = (c < EDIM) ? f2bf(e[c]) : (unsigned short)0;
}

// ---------------------------------------------------------------------------
// One GRU timestep: per WG 32 rows.
//  A(LDS) = [x | h]   -> gates = sigmoid(A @ WgT + bg)  (WMMA, 2x4 blocked)
//  A h-part <- r*h    -> cand  = tanh(A @ WcT + bc)     (WMMA, 2x2 blocked)
//  h = u*h + (1-u)*c with length mask; masked output scattered into ENC
//  (bwd rows write to their reverse_sequence slot, cols 256..511).
// ---------------------------------------------------------------------------
__global__ void __launch_bounds__(256)
k_gru_step(const unsigned short* __restrict__ X,
           const unsigned short* __restrict__ WgT,
           const unsigned short* __restrict__ WcT,
           const float* __restrict__ bg, const float* __restrict__ bc,
           float* __restrict__ Hst, unsigned short* __restrict__ ENC,
           const int* __restrict__ lens, int t) {
    extern __shared__ __align__(16) char smem[];
    unsigned short* A = (unsigned short*)smem;     // MT*LDA halves
    unsigned short* R = A + MT * LDA;              // MT*HDIM halves (r*h)
    unsigned short* U = R + MT * HDIM;             // MT*HDIM halves (update gate)

    const int tid  = threadIdx.x;
    const int lane = tid & 31;
    const int wv   = tid >> 5;
    const int m0   = blockIdx.x * MT;
    const int nlo  = lane & 15;
    const int rhi  = (lane >> 4) << 3;

    // ---- stage A = [x | h | 0] ----
    {
        const unsigned short* Xt = X + ((size_t)t * N2 + m0) * XS;
        // x part: async global -> LDS, 16B granules (ASYNCcnt)
        for (int i = tid; i < MT * (XS / 8); i += 256) {
            int row = i / (XS / 8), c = (i % (XS / 8)) * 8;
            unsigned loff = (unsigned)((row * LDA + c) * 2);
            async_g2l_b128(loff, (const void*)&Xt[(size_t)row * XS + c]);
        }
        // h part: fp32 -> bf16 via VGPRs
        for (int i = tid; i < MT * HDIM; i += 256) {
            int row = i >> 8, c = i & 255;
            A[row * LDA + XS + c] = f2bf(Hst[(size_t)(m0 + row) * HDIM + c]);
        }
        for (int i = tid; i < MT * 16; i += 256)
            A[(i >> 4) * LDA + XS + HDIM + (i & 15)] = 0;
        wait_async0();
    }
    __syncthreads();

    // ---- gates GEMM: 32x512 output; wave = 2 M-tiles x 4 N-tiles ----
    {
        f32x8 acc[2][4];
        #pragma unroll
        for (int j = 0; j < 4; ++j) {
            float b0 = bg[(wv * 4 + j) * 16 + nlo];
            #pragma unroll
            for (int v = 0; v < 8; ++v) { acc[0][j][v] = b0; acc[1][j][v] = b0; }
        }
        #pragma unroll
        for (int kt = 0; kt < KPAD / 32; ++kt) {
            bf16x16 a0 = load_frag(A, LDA, 0,  kt * 32, lane);
            bf16x16 a1 = load_frag(A, LDA, 16, kt * 32, lane);
            #pragma unroll
            for (int j = 0; j < 4; ++j) {
                bf16x16 b = load_frag(WgT, KPAD, (wv * 4 + j) * 16, kt * 32, lane);
                acc[0][j] = wmma_bf16(a0, b, acc[0][j]);
                acc[1][j] = wmma_bf16(a1, b, acc[1][j]);
            }
        }
        #pragma unroll
        for (int mi = 0; mi < 2; ++mi) {
            #pragma unroll
            for (int j = 0; j < 4; ++j) {
                int n = (wv * 4 + j) * 16 + nlo;
                #pragma unroll
                for (int v = 0; v < 8; ++v) {
                    int row = mi * 16 + rhi + v;
                    float g = 1.f / (1.f + __expf(-acc[mi][j][v]));
                    if (n < HDIM) {                       // r gate -> r*h
                        float h = bf2f(A[row * LDA + XS + n]);
                        R[row * HDIM + n] = f2bf(g * h);
                    } else {                              // u gate
                        U[row * HDIM + (n - HDIM)] = f2bf(g);
                    }
                }
            }
        }
    }
    __syncthreads();

    // ---- A h-part <- r*h ----
    for (int i = tid; i < MT * (HDIM / 8); i += 256) {
        int row = i / (HDIM / 8), c = (i % (HDIM / 8)) * 8;
        *(u16x8*)&A[row * LDA + XS + c] = *(const u16x8*)&R[row * HDIM + c];
    }
    __syncthreads();

    // ---- candidate GEMM: 32x256; wave = 2 M-tiles x 2 N-tiles; update ----
    {
        f32x8 acc[2][2];
        #pragma unroll
        for (int j = 0; j < 2; ++j) {
            float b0 = bc[(wv * 2 + j) * 16 + nlo];
            #pragma unroll
            for (int v = 0; v < 8; ++v) { acc[0][j][v] = b0; acc[1][j][v] = b0; }
        }
        #pragma unroll
        for (int kt = 0; kt < KPAD / 32; ++kt) {
            bf16x16 a0 = load_frag(A, LDA, 0,  kt * 32, lane);
            bf16x16 a1 = load_frag(A, LDA, 16, kt * 32, lane);
            #pragma unroll
            for (int j = 0; j < 2; ++j) {
                bf16x16 b = load_frag(WcT, KPAD, (wv * 2 + j) * 16, kt * 32, lane);
                acc[0][j] = wmma_bf16(a0, b, acc[0][j]);
                acc[1][j] = wmma_bf16(a1, b, acc[1][j]);
            }
        }
        #pragma unroll
        for (int mi = 0; mi < 2; ++mi) {
            #pragma unroll
            for (int j = 0; j < 2; ++j) {
                int n = (wv * 2 + j) * 16 + nlo;
                #pragma unroll
                for (int v = 0; v < 8; ++v) {
                    int row  = mi * 16 + rhi + v;
                    int grow = m0 + row;
                    float c    = tanhf(acc[mi][j][v]);
                    float u    = bf2f(U[row * HDIM + n]);
                    float hold = Hst[(size_t)grow * HDIM + n];
                    float hnew = u * hold + (1.f - u) * c;
                    int seq = (grow < NSEQ) ? grow : grow - NSEQ;
                    int len = lens[seq];
                    bool live = (t < len);
                    Hst[(size_t)grow * HDIM + n] = live ? hnew : hold;
                    unsigned short ov = f2bf(live ? hnew : 0.f);
                    size_t pos;
                    if (grow < NSEQ) {
                        pos = ((size_t)seq * TSEQ + t) * H2 + n;
                    } else {
                        int tp = live ? (len - 1 - t) : t;
                        pos = ((size_t)seq * TSEQ + tp) * H2 + HDIM + n;
                    }
                    ENC[pos] = ov;
                }
            }
        }
    }
}

// ---------------------------------------------------------------------------
// Projection: PROJ = tanh(ENC[64000x512] @ WpT + bp)  (WMMA, 2x2 blocked;
// ENC rows + Wp are L2-resident, no LDS staging needed).
// ---------------------------------------------------------------------------
__global__ void __launch_bounds__(256)
k_proj(const unsigned short* __restrict__ ENC, const unsigned short* __restrict__ WpT,
       const float* __restrict__ bp, float* __restrict__ PROJ) {
    int m0   = blockIdx.x * MT;
    int tid  = threadIdx.x, lane = tid & 31, wv = tid >> 5;
    const int nlo = lane & 15, rhi = (lane >> 4) << 3;

    f32x8 acc[2][2];
    #pragma unroll
    for (int j = 0; j < 2; ++j) {
        float b0 = bp[(wv * 2 + j) * 16 + nlo];
        #pragma unroll
        for (int v = 0; v < 8; ++v) { acc[0][j][v] = b0; acc[1][j][v] = b0; }
    }
    #pragma unroll
    for (int kt = 0; kt < H2 / 32; ++kt) {
        bf16x16 a0 = load_frag(ENC, H2, m0,      kt * 32, lane);
        bf16x16 a1 = load_frag(ENC, H2, m0 + 16, kt * 32, lane);
        #pragma unroll
        for (int j = 0; j < 2; ++j) {
            bf16x16 b = load_frag(WpT, H2, (wv * 2 + j) * 16, kt * 32, lane);
            acc[0][j] = wmma_bf16(a0, b, acc[0][j]);
            acc[1][j] = wmma_bf16(a1, b, acc[1][j]);
        }
    }
    #pragma unroll
    for (int mi = 0; mi < 2; ++mi) {
        #pragma unroll
        for (int j = 0; j < 2; ++j) {
            int n = (wv * 2 + j) * 16 + nlo;
            #pragma unroll
            for (int v = 0; v < 8; ++v) {
                int row = m0 + mi * 16 + rhi + v;
                PROJ[(size_t)row * HDIM + n] = tanhf(acc[mi][j][v]);
            }
        }
    }
}

// ---------------------------------------------------------------------------
// Attention pooling per sequence: scores via wave dot-reduce, softmax over T,
// weighted sum of PROJ rows.
// ---------------------------------------------------------------------------
__global__ void k_attn(const float* __restrict__ PROJ, const float* __restrict__ ctx,
                       float* __restrict__ out) {
    __shared__ float sc[TSEQ];
    int n = blockIdx.x;
    int tid = threadIdx.x, lane = tid & 31, wv = tid >> 5;
    const float* P = PROJ + (size_t)n * TSEQ * HDIM;

    for (int t = wv; t < TSEQ; t += 8) {
        float s = 0.f;
        for (int c = lane; c < HDIM; c += 32) s += P[t * HDIM + c] * ctx[c];
        #pragma unroll
        for (int off = 16; off > 0; off >>= 1) s += __shfl_xor(s, off, 32);
        if (lane == 0) sc[t] = s;
    }
    __syncthreads();

    float mx = -1e30f;
    for (int t = 0; t < TSEQ; ++t) mx = fmaxf(mx, sc[t]);
    float denom = 0.f;
    for (int t = 0; t < TSEQ; ++t) denom += __expf(sc[t] - mx);

    int c = tid;                                   // 256 threads = HDIM cols
    float o = 0.f;
    for (int t = 0; t < TSEQ; ++t) o += __expf(sc[t] - mx) * P[t * HDIM + c];
    out[(size_t)n * HDIM + c] = o / denom;
}

// ---------------------------------------------------------------------------
extern "C" void kernel_launch(void* const* d_in, const int* in_sizes, int n_in,
                              void* d_out, int out_size, void* d_ws, size_t ws_size,
                              hipStream_t stream) {
    (void)in_sizes; (void)n_in; (void)out_size; (void)ws_size;
    const int*   ids  = (const int*)d_in[0];
    const int*   lens = (const int*)d_in[1];
    const float* Emb  = (const float*)d_in[2];
    const float* Wg   = (const float*)d_in[3];
    const float* bg   = (const float*)d_in[4];
    const float* Wc   = (const float*)d_in[5];
    const float* bc   = (const float*)d_in[6];
    const float* Wp   = (const float*)d_in[7];
    const float* bp   = (const float*)d_in[8];
    const float* ctx  = (const float*)d_in[9];
    float* out = (float*)d_out;

    char* ws = (char*)d_ws;
    size_t off = 0;
    auto alloc = [&](size_t bytes) -> char* {
        char* p = ws + off;
        off = (off + bytes + 255) & ~(size_t)255;
        return p;
    };
    unsigned short* X   = (unsigned short*)alloc((size_t)TSEQ * N2 * XS * 2);
    unsigned short* WgT = (unsigned short*)alloc((size_t)H2 * KPAD * 2);
    unsigned short* WcT = (unsigned short*)alloc((size_t)HDIM * KPAD * 2);
    unsigned short* WpT = (unsigned short*)alloc((size_t)HDIM * H2 * 2);
    float*          Hst = (float*)alloc((size_t)N2 * HDIM * 4);
    unsigned short* ENC = (unsigned short*)alloc((size_t)NSEQ * TSEQ * H2 * 2);
    float*          PROJ= (float*)alloc((size_t)NSEQ * TSEQ * HDIM * 4);

    hipMemsetAsync(Hst, 0, (size_t)N2 * HDIM * 4, stream);

    int packTot = H2 * KPAD + HDIM * KPAD + HDIM * H2;
    k_pack_w<<<(packTot + 255) / 256, 256, 0, stream>>>(Wg, Wc, Wp, WgT, WcT, WpT);
    k_gather<<<TSEQ * N2, 128, 0, stream>>>(ids, lens, Emb, X);

    size_t smem = (size_t)(MT * LDA + 2 * MT * HDIM) * 2;   // ~70 KB dynamic LDS
    for (int t = 0; t < TSEQ; ++t)
        k_gru_step<<<N2 / MT, 256, smem, stream>>>(X, WgT, WcT, bg, bc, Hst, ENC, lens, t);

    k_proj<<<(NSEQ * TSEQ) / MT, 256, 0, stream>>>(ENC, WpT, bp, PROJ);
    k_attn<<<NSEQ, 256, 0, stream>>>(PROJ, ctx, out);
}
